// gs_block_41558103556339
// MI455X (gfx1250) — matrix operand order repaired and checked by reading, not compiled, not run on hardware
//
#include <hip/hip_runtime.h>
#include <hip/hip_bf16.h>

// ---------------------------------------------------------------------------
// GraphSAGE-style block for MI455X (gfx1250, wave32, WMMA + TDM + async-LDS)
//   neigh = rownorm(Adj - I) @ x          (dense 8192x8192x512, bf16 WMMA)
//   out   = l2norm_rows(relu([x|neigh] @ W^T))   (8192x1024x512, bf16 WMMA)
// ---------------------------------------------------------------------------

#define N_NODES 8192
#define FEAT    512
#define EMBED   512
#define KCAT    (2 * FEAT)

typedef __attribute__((ext_vector_type(16))) __bf16       v16bf;
typedef __attribute__((ext_vector_type(8)))  float        v8f;
typedef __attribute__((ext_vector_type(4)))  unsigned int u32x4;
typedef __attribute__((ext_vector_type(8)))  unsigned int u32x8;
typedef __attribute__((ext_vector_type(4)))  float        f32x4;

union Frag16 { u32x4 q[2]; v16bf v; };   // 32B: one 16x32 bf16 A-frag / 32x16 B-frag per lane

__device__ __forceinline__ unsigned short f2bf_bits(float f) {
  return __builtin_bit_cast(unsigned short, (__bf16)f);
}

// 32-bit LDS byte address of a generic pointer to __shared__ (low 32 bits).
__device__ __forceinline__ unsigned lds_addr_of(const void* p) {
  return (unsigned)(size_t)p;
}

// ---- CDNA5 async global->LDS copies (ASYNCcnt-tracked, GVS addressing) ----
__device__ __forceinline__ void async_ld_b128(unsigned lds_addr, unsigned voff,
                                              unsigned long long sbase) {
  asm volatile("global_load_async_to_lds_b128 %0, %1, %2"
               :
               : "v"(lds_addr), "v"(voff), "s"(sbase)
               : "memory");
}
__device__ __forceinline__ void async_ld_b64(unsigned lds_addr, unsigned voff,
                                             unsigned long long sbase) {
  asm volatile("global_load_async_to_lds_b64 %0, %1, %2"
               :
               : "v"(lds_addr), "v"(voff), "s"(sbase)
               : "memory");
}
__device__ __forceinline__ void wait_async0() {
  asm volatile("s_wait_asynccnt 0x0" ::: "memory");
}

// ---- CDNA5 Tensor Data Mover: 2D tile global->LDS (TENSORcnt-tracked) ----
// D# per ISA cdna5 ch.8: group0 = {count, lds_addr, global_addr, type=2},
// group1 = {data_size, pad cfg, tensor dims, tile dims, dim0 stride}.
// 16-bit elements; optional LDS row padding (pad after every `padIntLog2`
// 8-byte units, insert (padAmt+1) DWORDs).
__device__ __forceinline__ void tdm_load_2d_bf16(
    unsigned lds_addr, unsigned long long gaddr, unsigned tile_d0,
    unsigned tile_d1, unsigned long long tensor_d0, unsigned tensor_d1,
    unsigned long long stride0, unsigned padInterval, unsigned padAmount,
    bool padEn) {
  u32x4 g0;
  g0[0] = 1u;                                            // count=1 (valid)
  g0[1] = lds_addr;                                      // lds_addr
  g0[2] = (unsigned)(gaddr & 0xFFFFFFFFull);             // global_addr[31:0]
  g0[3] = (unsigned)((gaddr >> 32) & 0x01FFFFFFull)      // global_addr[56:32]
          | (2u << 30);                                  // type=2 ("image")
  u32x8 g1;
  g1[0] = (1u << 16)                                     // data_size=1 -> 2B
          | (padEn ? ((1u << 20) | (padInterval << 22) | (padAmount << 25))
                   : 0u);
  g1[1] = (unsigned)((tensor_d0 & 0xFFFFull) << 16);     // tensor_dim0[15:0]
  g1[2] = (unsigned)((tensor_d0 >> 16) & 0xFFFFull)      // tensor_dim0[31:16]
          | ((tensor_d1 & 0xFFFFu) << 16);               // tensor_dim1[15:0]
  g1[3] = ((tensor_d1 >> 16) & 0xFFFFu)                  // tensor_dim1[31:16]
          | (tile_d0 << 16);                             // tile_dim0
  g1[4] = tile_d1 & 0xFFFFu;                             // tile_dim1 (d2=0)
  g1[5] = (unsigned)(stride0 & 0xFFFFFFFFull);           // dim0_stride[31:0]
  g1[6] = (unsigned)((stride0 >> 32) & 0xFFFFull);       // dim0_stride[47:32]
  g1[7] = 0u;                                            // dim1_stride=0
  asm volatile("tensor_load_to_lds %0, %1"
               :
               : "s"(g0), "s"(g1)
               : "memory");
}
__device__ __forceinline__ void wait_tensor0() {
  __builtin_amdgcn_s_wait_tensorcnt(0);
}

// --------------------------- elementwise f32 -> bf16 -----------------------
__global__ void k_f32_to_bf16(const float* __restrict__ src,
                              __bf16* __restrict__ dst, int n) {
  int i = blockIdx.x * blockDim.x + threadIdx.x;
  if (i < n) dst[i] = (__bf16)src[i];
}

// --------------------- x (f32 [N][F]) -> xT (bf16 [F][N]) ------------------
__global__ __launch_bounds__(256) void k_transpose_bf16(
    const float* __restrict__ x, __bf16* __restrict__ xT) {
  __shared__ float tile[32][33];
  const int i0 = blockIdx.x * 32;              // node dim
  const int j0 = blockIdx.y * 32;              // feat dim
  const int lx = threadIdx.x & 31;
  const int ly = threadIdx.x >> 5;             // 0..7
#pragma unroll
  for (int r = 0; r < 4; ++r) {
    int row = ly + r * 8;
    tile[row][lx] = x[(size_t)(i0 + row) * FEAT + j0 + lx];
  }
  __syncthreads();
#pragma unroll
  for (int r = 0; r < 4; ++r) {
    int row = ly + r * 8;                      // feat-local
    xT[(size_t)(j0 + row) * N_NODES + i0 + lx] = (__bf16)tile[lx][row];
  }
}

// ---------------------------------------------------------------------------
// Kernel: fused  [row-degree  +  Adj @ x  +  (S - x)/deg]  ->  neigh (bf16)
// Block: 256 thr (8 waves), tile BM=64 x BN=512 (full feat width) -> Adj read
// exactly once (~11 us HBM floor). Wave grid 2x4, wave tile 32x128 = 16 acc
// tiles. A stage: sync VALU (fused f32->bf16 + degree). B stage: one TDM
// descriptor per K-step (wave 0 issues; DMA pads LDS rows to 80B pitch).
// ---------------------------------------------------------------------------
__global__ __launch_bounds__(256) void k_aggregate(
    const float* __restrict__ Adj, const float* __restrict__ x,
    const __bf16* __restrict__ xT, __bf16* __restrict__ neigh) {
  __shared__ __align__(16) unsigned short As[64][40];    // 64x32 bf16 (+pad)
  __shared__ __align__(16) unsigned short Bs[512][40];   // Bt[n][k] (+pad)
  __shared__ float partial[256];
  __shared__ float invN[64];

  const int t     = threadIdx.x;
  const int lane  = t & 31;
  const int wid   = t >> 5;
  const int waveM = wid & 1;                  // 0..1 -> 32 rows each
  const int waveN = wid >> 1;                 // 0..3 -> 128 cols each
  const int half  = lane >> 4;
  const int l15   = lane & 15;
  const int row0  = blockIdx.x * 64;

  const int ar = t >> 2;                      // A-stage row 0..63
  const int ac = (t & 3) * 8;                 // A-stage col chunk

  const unsigned long long xTbase = (unsigned long long)(size_t)xT;
  const unsigned bsBase = lds_addr_of(&Bs[0][0]);

  v8f acc[16];
#pragma unroll
  for (int i = 0; i < 16; ++i)
#pragma unroll
    for (int v = 0; v < 8; ++v) acc[i][v] = 0.0f;
  float rowsum = 0.0f;

  for (int k0 = 0; k0 < N_NODES; k0 += 32) {
    __syncthreads();
    // ---- stage B (TDM): Bt[n][0:32] = xT[n][k0:k0+32], 512 rows x 64B,
    //      hardware-padded to 80B LDS pitch (interval 16 DW, amount 4 DW).
    if (wid == 0) {
      tdm_load_2d_bf16(bsBase, xTbase + (unsigned long long)k0 * 2,
                       /*tile_d0=*/32, /*tile_d1=*/512,
                       /*tensor_d0=*/N_NODES, /*tensor_d1=*/FEAT,
                       /*stride0=*/N_NODES,
                       /*padInterval=*/3, /*padAmount=*/3, /*padEn=*/true);
    }
    // ---- stage A (sync): Adj f32 -> bf16 LDS, fused per-row degree accum
    {
      const f32x4* gp =
          (const f32x4*)&Adj[(size_t)(row0 + ar) * N_NODES + k0 + ac];
      f32x4 f0 = gp[0];
      f32x4 f1 = gp[1];
      rowsum += f0.x + f0.y + f0.z + f0.w + f1.x + f1.y + f1.z + f1.w;
      unsigned int* ws = (unsigned int*)&As[ar][ac];
      ws[0] = (unsigned)f2bf_bits(f0.x) | ((unsigned)f2bf_bits(f0.y) << 16);
      ws[1] = (unsigned)f2bf_bits(f0.z) | ((unsigned)f2bf_bits(f0.w) << 16);
      ws[2] = (unsigned)f2bf_bits(f1.x) | ((unsigned)f2bf_bits(f1.y) << 16);
      ws[3] = (unsigned)f2bf_bits(f1.z) | ((unsigned)f2bf_bits(f1.w) << 16);
      if (k0 + 32 < N_NODES)  // keep the 23.3 TB/s Adj stream ahead of compute
        __builtin_prefetch(&Adj[(size_t)(row0 + ar) * N_NODES + k0 + 32 + ac],
                           0, 1);
    }
    if (wid == 0) wait_tensor0();   // issuing wave gates barrier on DMA done
    __syncthreads();
    // ---- compute: 16 v_wmma_f32_16x16x32_bf16 per wave per K-step
    Frag16 a0, a1;
    {
      int m0 = waveM * 32 + l15;
      a0.q[0] = *(const u32x4*)&As[m0][half * 8];
      a0.q[1] = *(const u32x4*)&As[m0][16 + half * 8];
      int m1 = m0 + 16;
      a1.q[0] = *(const u32x4*)&As[m1][half * 8];
      a1.q[1] = *(const u32x4*)&As[m1][16 + half * 8];
    }
#pragma unroll
    for (int tn = 0; tn < 8; ++tn) {
      Frag16 b;
      int n = waveN * 128 + tn * 16 + l15;
      const u32x4* bp = (const u32x4*)&Bs[n][half * 8];
      b.q[0] = bp[0]; b.q[1] = bp[2];
      acc[tn * 2 + 0] = __builtin_amdgcn_wmma_f32_16x16x32_bf16(
          false, a0.v, false, b.v, (short)0, acc[tn * 2 + 0], false, false);
      acc[tn * 2 + 1] = __builtin_amdgcn_wmma_f32_16x16x32_bf16(
          false, a1.v, false, b.v, (short)0, acc[tn * 2 + 1], false, false);
    }
  }

  // ---- reduce per-row degree, invert (self-loop removed: deg = rowsum - 1)
  partial[t] = rowsum;
  __syncthreads();
  if (t < 64) {
    float s = partial[t * 4] + partial[t * 4 + 1] + partial[t * 4 + 2] +
              partial[t * 4 + 3];
    invN[t] = 1.0f / (s - 1.0f);
  }
  __syncthreads();

  // ---- epilogue: neigh = (S - x[i]) * invN  (f32 math, store bf16)
  unsigned short* nb = (unsigned short*)neigh;
#pragma unroll
  for (int tm = 0; tm < 2; ++tm) {
#pragma unroll
    for (int tn = 0; tn < 8; ++tn) {
      v8f c = acc[tn * 2 + tm];
#pragma unroll
      for (int v = 0; v < 8; ++v) {
        int m = waveM * 32 + tm * 16 + v + 8 * half;
        int gm = row0 + m;
        int n = waveN * 128 + tn * 16 + l15;
        float val = (c[v] - x[(size_t)gm * FEAT + n]) * invN[m];
        nb[(size_t)gm * FEAT + n] = f2bf_bits(val);
      }
    }
  }
}

// ---------------------------------------------------------------------------
// Kernel: fused  out = l2norm_rows(relu([x|neigh] @ W^T))
// Block: 256 thr, tile BM=32 x BN=512 (full embed width -> row norm in LDS).
// Wave grid 2x4, wave tile 16x128 = 8 acc tiles. Both A and B stages are pure
// bf16 copies -> async-to-LDS (B = W^T transposed is weight's native layout).
// ---------------------------------------------------------------------------
__global__ __launch_bounds__(256) void k_linear_norm(
    const __bf16* __restrict__ xbf, const __bf16* __restrict__ neighbf,
    const __bf16* __restrict__ wbf, float* __restrict__ out) {
  __shared__ __align__(16) unsigned short As[32][40];
  __shared__ __align__(16) unsigned short Bs[512][40];
  __shared__ float outT[32][520];
  __shared__ float scale[32];

  const int t     = threadIdx.x;
  const int lane  = t & 31;
  const int wid   = t >> 5;
  const int waveM = wid & 1;                  // 16 rows each
  const int waveN = wid >> 1;                 // 128 cols each
  const int half  = lane >> 4;
  const int l15   = lane & 15;
  const int row0  = blockIdx.x * 32;

  const unsigned long long xbase = (unsigned long long)(size_t)xbf;
  const unsigned long long nbase = (unsigned long long)(size_t)neighbf;
  const unsigned long long wbase = (unsigned long long)(size_t)wbf;
  const unsigned asBase = lds_addr_of(&As[0][0]);
  const unsigned bsBase = lds_addr_of(&Bs[0][0]);

  const int arow = t >> 3;                    // A-stage row 0..31
  const int aj   = t & 7;                     // A-stage 8B chunk
  const int brow = t >> 2;                    // B-stage row within 64-row group
  const int bj   = t & 3;                     // B-stage 16B chunk

  v8f acc[8];
#pragma unroll
  for (int i = 0; i < 8; ++i)
#pragma unroll
    for (int v = 0; v < 8; ++v) acc[i][v] = 0.0f;

  for (int k0 = 0; k0 < KCAT; k0 += 32) {
    __syncthreads();
    // ---- stage A (async): combined = [x | neigh] rows; 32 rows x 64B
    {
      unsigned lds = asBase + (unsigned)(arow * 80 + aj * 8);
      if (k0 < FEAT) {
        unsigned voff =
            (unsigned)((row0 + arow) * (FEAT * 2) + k0 * 2 + aj * 8);
        async_ld_b64(lds, voff, xbase);
      } else {
        unsigned voff =
            (unsigned)((row0 + arow) * (FEAT * 2) + (k0 - FEAT) * 2 + aj * 8);
        async_ld_b64(lds, voff, nbase);
      }
    }
    // ---- stage B (async): Bt[e][k] = weight[e][k0+k]; 512 rows x 64B
    {
#pragma unroll
      for (int it = 0; it < 8; ++it) {
        int e = it * 64 + brow;
        unsigned lds = bsBase + (unsigned)(e * 80 + bj * 16);
        unsigned voff = (unsigned)(e * (KCAT * 2) + k0 * 2 + bj * 16);
        async_ld_b128(lds, voff, wbase);
      }
    }
    wait_async0();
    __syncthreads();
    Frag16 a;
    {
      int m = waveM * 16 + l15;
      a.q[0] = *(const u32x4*)&As[m][half * 8];
      a.q[1] = *(const u32x4*)&As[m][16 + half * 8];
    }
#pragma unroll
    for (int tn = 0; tn < 8; ++tn) {
      Frag16 b;
      int n = waveN * 128 + tn * 16 + l15;
      const u32x4* bp = (const u32x4*)&Bs[n][half * 8];
      b.q[0] = bp[0]; b.q[1] = bp[2];
      acc[tn] = __builtin_amdgcn_wmma_f32_16x16x32_bf16(
          false, a.v, false, b.v, (short)0, acc[tn], false, false);
    }
  }
  __syncthreads();

  // ---- ReLU into LDS strip (each block owns complete rows)
#pragma unroll
  for (int tn = 0; tn < 8; ++tn) {
    v8f c = acc[tn];
#pragma unroll
    for (int v = 0; v < 8; ++v) {
      int m = waveM * 16 + v + 8 * half;
      int n = waveN * 128 + tn * 16 + l15;
      outT[m][n] = fmaxf(c[v], 0.0f);
    }
  }
  __syncthreads();
  // ---- per-row L2 norm (F.normalize: out / max(||out||, eps))
  if (t < 32) {
    float s = 0.0f;
    for (int n = 0; n < EMBED; ++n) {
      float v = outT[t][n];
      s += v * v;
    }
    scale[t] = 1.0f / fmaxf(sqrtf(s), 1e-12f);
  }
  __syncthreads();
  // ---- scaled f32 store, 64 elems/thread as float4s
  {
    int r  = t >> 3;
    int c0 = (t & 7) * 64;
    float sc = scale[r];
    f32x4* op = (f32x4*)&out[(size_t)(row0 + r) * EMBED + c0];
#pragma unroll
    for (int q = 0; q < 16; ++q) {
      f32x4 v4;
      v4.x = outT[r][c0 + q * 4 + 0] * sc;
      v4.y = outT[r][c0 + q * 4 + 1] * sc;
      v4.z = outT[r][c0 + q * 4 + 2] * sc;
      v4.w = outT[r][c0 + q * 4 + 3] * sc;
      op[q] = v4;
    }
  }
}

// ---------------------------------------------------------------------------
extern "C" void kernel_launch(void* const* d_in, const int* in_sizes, int n_in,
                              void* d_out, int out_size, void* d_ws,
                              size_t ws_size, hipStream_t stream) {
  const float* x      = (const float*)d_in[0];  // [8192, 512]
  const float* Adj    = (const float*)d_in[1];  // [8192, 8192]
  const float* weight = (const float*)d_in[2];  // [512, 1024]
  float* out = (float*)d_out;                   // [8192, 512]

  char* ws = (char*)d_ws;
  __bf16* xbf  = (__bf16*)(ws);                             //  8 MiB [N][F]
  __bf16* xTbf = (__bf16*)(ws + (size_t)8  * 1024 * 1024);  //  8 MiB [F][N]
  __bf16* wbf  = (__bf16*)(ws + (size_t)16 * 1024 * 1024);  //  1 MiB [E][2F]
  __bf16* nbf  = (__bf16*)(ws + (size_t)17 * 1024 * 1024);  //  8 MiB [N][F]

  k_f32_to_bf16<<<(N_NODES * FEAT) / 256, 256, 0, stream>>>(x, xbf,
                                                            N_NODES * FEAT);
  k_f32_to_bf16<<<(EMBED * KCAT) / 256, 256, 0, stream>>>(weight, wbf,
                                                          EMBED * KCAT);
  k_transpose_bf16<<<dim3(N_NODES / 32, FEAT / 32), 256, 0, stream>>>(x, xTbf);
  k_aggregate<<<N_NODES / 64, 256, 0, stream>>>(Adj, x, xTbf, nbf);
  k_linear_norm<<<N_NODES / 32, 256, 0, stream>>>(xbf, nbf, wbf, out);
}
